// Mamba_decoder_37744172597986
// MI455X (gfx1250) — compile-verified, matrix operand order checked by reading, not run on hardware
//
#include <hip/hip_runtime.h>
#include <hip/hip_bf16.h>

typedef __attribute__((ext_vector_type(16))) __bf16 v16bf;
typedef __attribute__((ext_vector_type(8)))  float  v8f;
typedef __attribute__((ext_vector_type(8)))  unsigned int v8u;

#define BTOK 32768
#define DM 256
#define DS 16
#define DI 256
#define DTR 16
#define FF 512
#define TOK 16   // tokens per block

union B16x16 { v8u u; v16bf b; };

__device__ __forceinline__ unsigned short f2bf(float x) {
    unsigned int u = __float_as_uint(x);
    unsigned int r = u + 0x7FFFu + ((u >> 16) & 1u);
    return (unsigned short)(r >> 16);
}
__device__ __forceinline__ unsigned int pack2bf(float a, float b) {
    return (unsigned int)f2bf(a) | ((unsigned int)f2bf(b) << 16);
}
__device__ __forceinline__ float softplusf(float x) {
    return (x > 20.f) ? x : log1pf(__expf(x));
}
__device__ __forceinline__ float siluf(float x) {
    return x / (1.f + __expf(-x));
}
__device__ __forceinline__ float sigmoidf(float x) {
    return 1.f / (1.f + __expf(-x));
}

__device__ __forceinline__ v8f wmma_bf16(v16bf a, v16bf b, v8f c) {
    return __builtin_amdgcn_wmma_f32_16x16x32_bf16(
        false, a, false, b, (short)0, c, false, false);
}

// A fragment (16x32 bf16, M rows in lanes, K striped per ISA 7.12.2)
// base: LDS viewed as dwords (bf16 pairs), stride_dw = row stride in dwords
__device__ __forceinline__ v16bf load_a_frag(const unsigned int* base,
                                             int stride_dw, int kbase, int lane) {
    int m = lane & 15;
    int h = (lane >> 4) << 3;
    const unsigned int* row = base + m * stride_dw;
    B16x16 t;
#pragma unroll
    for (int v = 0; v < 8; ++v) {
        int k = kbase + ((v >> 2) << 4) + h + ((v & 3) << 1);
        t.u[v] = row[k >> 1];
    }
    return t.b;
}

// B fragment pre-packed in workspace: 8 contiguous dwords per (frag, lane)
__device__ __forceinline__ v16bf load_b_frag(const unsigned int* packed,
                                             int frag, int lane) {
    const uint4* p = (const uint4*)(packed + (((size_t)frag * 32 + lane) << 3));
    uint4 lo = p[0];
    uint4 hi = p[1];
    B16x16 t;
    t.u[0] = lo.x; t.u[1] = lo.y; t.u[2] = lo.z; t.u[3] = lo.w;
    t.u[4] = hi.x; t.u[5] = hi.y; t.u[6] = hi.z; t.u[7] = hi.w;
    return t.b;
}

// ---- weight packing: W is (N, K) row-major fp32; emit bf16 B fragments ----
__global__ __launch_bounds__(256) void pack_wb(const float* __restrict__ W,
                                               int N, int K, int ksteps, int ntiles,
                                               unsigned int* __restrict__ out) {
    int idx = blockIdx.x * blockDim.x + threadIdx.x;
    int total = ntiles * ksteps * 32;
    if (idx >= total) return;
    int lane = idx & 31;
    int fs   = idx >> 5;
    int s    = fs % ksteps;
    int j    = fs / ksteps;
    int n    = j * 16 + (lane & 15);
    int h    = (lane >> 4) << 3;
    unsigned int frag[8];
#pragma unroll
    for (int v = 0; v < 8; ++v) {
        int k = s * 32 + ((v >> 2) << 4) + h + ((v & 3) << 1);
        float a = (n < N && k     < K) ? W[(size_t)n * K + k]     : 0.f;
        float b = (n < N && (k+1) < K) ? W[(size_t)n * K + k + 1] : 0.f;
        frag[v] = pack2bf(a, b);
    }
    uint4* o = (uint4*)(out + ((size_t)idx << 3));
    o[0] = make_uint4(frag[0], frag[1], frag[2], frag[3]);
    o[1] = make_uint4(frag[4], frag[5], frag[6], frag[7]);
}

// ---- fused mamba decoder step, 16 tokens per block ----
__global__ __launch_bounds__(256) void mamba_main(
    const float* __restrict__ x0, const float* __restrict__ ff,
    const float* __restrict__ h,
    const float* __restrict__ b_dt, const float* __restrict__ A_log,
    const float* __restrict__ Dv,
    const float* __restrict__ b1, const float* __restrict__ b2,
    const float* __restrict__ gamma, const float* __restrict__ beta,
    const float* __restrict__ b_bb,
    const unsigned int* __restrict__ pWin, const unsigned int* __restrict__ pWx,
    const unsigned int* __restrict__ pWdt, const unsigned int* __restrict__ pW1,
    const unsigned int* __restrict__ pW2,  const unsigned int* __restrict__ pWbb,
    float* __restrict__ out, float* __restrict__ hnew) {

    __shared__ __align__(16) __hip_bfloat16 sA0[TOK * 256];  // A operand (K<=256)
    __shared__ __align__(16) __hip_bfloat16 sA1[TOK * 512];  // FFN hidden
    __shared__ float sU[TOK * 256];      // u, later s (post-FFN)
    __shared__ float sGate[TOK * 256];   // raw gate
    __shared__ float sDelta[TOK * 256];  // delta, later y
    __shared__ float sXdbl[TOK * 48];

    unsigned int* sA0d = (unsigned int*)sA0;
    const int tid  = threadIdx.x;
    const int w    = tid >> 5;
    const int lane = tid & 31;
    const int tok0 = blockIdx.x * TOK;

    // L2 prefetch hints for the shared packed weights (global_prefetch_b8)
    __builtin_prefetch((const char*)pW1 + ((size_t)tid << 6), 0, 0);
    __builtin_prefetch((const char*)pW2 + ((size_t)tid << 6), 0, 0);

    // ---- stage 1: sine positional embedding -> sA0 (16 x 256 bf16) ----
    for (int i = tid; i < TOK * 128; i += 256) {
        int row = i >> 7;
        int cp  = i & 127;
        float e[2];
#pragma unroll
        for (int q = 0; q < 2; ++q) {
            int col  = cp * 2 + q;
            int axis = col >> 6;
            int j    = col & 63;
            float xv = x0[(size_t)(tok0 + row) * 4 + axis];
            float dimt = __expf((float)(j >> 1) * (1.f / 32.f) * 9.210340371976184f);
            float ph = xv * 6.283185307179586f / dimt;
            e[q] = (j & 1) ? __cosf(ph) : __sinf(ph);
        }
        sA0d[row * 128 + cp] = pack2bf(e[0], e[1]);
    }
    __syncthreads();

    // ---- stage 2: pm = pos @ W_in^T (16x512); u = silu(pm[:,:256]); gate ----
    {
        v8f acc[4];
        acc[0] = {}; acc[1] = {}; acc[2] = {}; acc[3] = {};
#pragma unroll
        for (int s = 0; s < 8; ++s) {
            v16bf a = load_a_frag(sA0d, 128, s * 32, lane);
#pragma unroll
            for (int t = 0; t < 4; ++t) {
                v16bf b = load_b_frag(pWin, (w * 4 + t) * 8 + s, lane);
                acc[t] = wmma_bf16(a, b, acc[t]);
            }
        }
#pragma unroll
        for (int t = 0; t < 4; ++t) {
            int j   = w * 4 + t;
            int col = j * 16 + (lane & 15);
#pragma unroll
            for (int r = 0; r < 8; ++r) {
                int m = r + ((lane >> 4) << 3);
                float v = acc[t][r];
                if (j < 16) sU[m * 256 + col] = siluf(v);
                else        sGate[m * 256 + (col - 256)] = v;
            }
        }
    }
    __syncthreads();

    // ---- stage 3: flow_features tile -> sA0 bf16 ----
    for (int i = tid; i < TOK * 128; i += 256) {
        int row = i >> 7;
        int cp  = i & 127;
        const float* src = ff + (size_t)(tok0 + row) * 256 + cp * 2;
        sA0d[row * 128 + cp] = pack2bf(src[0], src[1]);
    }
    __syncthreads();

    // ---- stage 4: x_dbl = ff @ W_x^T (16x48) ----
    if (w < 3) {
        v8f acc = {};
#pragma unroll
        for (int s = 0; s < 8; ++s) {
            v16bf a = load_a_frag(sA0d, 128, s * 32, lane);
            v16bf b = load_b_frag(pWx, w * 8 + s, lane);
            acc = wmma_bf16(a, b, acc);
        }
        int col = w * 16 + (lane & 15);
#pragma unroll
        for (int r = 0; r < 8; ++r) {
            int m = r + ((lane >> 4) << 3);
            sXdbl[m * 48 + col] = acc[r];
        }
    }
    __syncthreads();

    // ---- stage 5: build dlt tile (16x32 bf16, K padded) into sA0 ----
    {
        int row = tid >> 4;
        int c   = tid & 15;
        unsigned int val = 0;
        if (c < 8) val = pack2bf(sXdbl[row * 48 + 2 * c], sXdbl[row * 48 + 2 * c + 1]);
        sA0d[row * 128 + c] = val;
    }
    __syncthreads();

    // ---- stage 6: delta = softplus(dlt @ W_dt^T + b_dt) (16x256) ----
    {
        v8f acc[2];
        acc[0] = {}; acc[1] = {};
        v16bf a = load_a_frag(sA0d, 128, 0, lane);
#pragma unroll
        for (int t = 0; t < 2; ++t) {
            v16bf b = load_b_frag(pWdt, w * 2 + t, lane);
            acc[t] = wmma_bf16(a, b, acc[t]);
        }
#pragma unroll
        for (int t = 0; t < 2; ++t) {
            int col = (w * 2 + t) * 16 + (lane & 15);
#pragma unroll
            for (int r = 0; r < 8; ++r) {
                int m = r + ((lane >> 4) << 3);
                sDelta[m * 256 + col] = softplusf(acc[t][r] + b_dt[col]);
            }
        }
    }
    __syncthreads();

    // ---- stage 7: SSM state update; h_new out; y = einsum + u*D, *silu(gate) ----
#pragma unroll 1
    for (int it = 0; it < 16; ++it) {
        int pi = it * 256 + tid;
        int t  = pi >> 8;
        int d  = pi & 255;
        float delta = sDelta[pi];
        float u     = sU[pi];
        float du    = delta * u;
        const float4* Arow = (const float4*)(A_log + d * 16);
        const float4* hrow = (const float4*)(h + ((size_t)(tok0 + t) * 256 + d) * 16);
        float4* hout = (float4*)(hnew + ((size_t)(tok0 + t) * 256 + d) * 16);
        const float* bm = &sXdbl[t * 48 + 16];
        const float* cm = &sXdbl[t * 48 + 32];
        float yacc = 0.f;
#pragma unroll
        for (int q = 0; q < 4; ++q) {
            float4 al = Arow[q];
            float4 hv = hrow[q];
            float4 r;
            r.x = __expf(delta * -__expf(al.x)) * hv.x + du * bm[q * 4 + 0];
            r.y = __expf(delta * -__expf(al.y)) * hv.y + du * bm[q * 4 + 1];
            r.z = __expf(delta * -__expf(al.z)) * hv.z + du * bm[q * 4 + 2];
            r.w = __expf(delta * -__expf(al.w)) * hv.w + du * bm[q * 4 + 3];
            yacc = fmaf(r.x, cm[q * 4 + 0], yacc);
            yacc = fmaf(r.y, cm[q * 4 + 1], yacc);
            yacc = fmaf(r.z, cm[q * 4 + 2], yacc);
            yacc = fmaf(r.w, cm[q * 4 + 3], yacc);
            hout[q] = r;
        }
        float gate = sGate[pi];
        float y = (yacc + u * Dv[d]) * siluf(gate);
        sDelta[pi] = y;                      // sDelta now holds y (fp32)
        sA0[pi]    = __float2bfloat16(y);    // bf16 A operand for FFN
    }
    __syncthreads();

    // ---- stage 8: hidden = relu(y @ W1^T + b1) (16x512 bf16 -> sA1) ----
    {
        v8f acc[4];
        acc[0] = {}; acc[1] = {}; acc[2] = {}; acc[3] = {};
#pragma unroll
        for (int s = 0; s < 8; ++s) {
            v16bf a = load_a_frag(sA0d, 128, s * 32, lane);
#pragma unroll
            for (int t = 0; t < 4; ++t) {
                v16bf b = load_b_frag(pW1, (w * 4 + t) * 8 + s, lane);
                acc[t] = wmma_bf16(a, b, acc[t]);
            }
        }
#pragma unroll
        for (int t = 0; t < 4; ++t) {
            int col = (w * 4 + t) * 16 + (lane & 15);
#pragma unroll
            for (int r = 0; r < 8; ++r) {
                int m = r + ((lane >> 4) << 3);
                float v = acc[t][r] + b1[col];
                sA1[m * 512 + col] = __float2bfloat16(v > 0.f ? v : 0.f);
            }
        }
    }
    __syncthreads();

    // ---- stage 9: s = y + hidden @ W2^T + b2  -> sU ----
    {
        v8f acc[2];
        acc[0] = {}; acc[1] = {};
#pragma unroll
        for (int s = 0; s < 16; ++s) {
            v16bf a = load_a_frag((const unsigned int*)sA1, 256, s * 32, lane);
#pragma unroll
            for (int t = 0; t < 2; ++t) {
                v16bf b = load_b_frag(pW2, (w * 2 + t) * 16 + s, lane);
                acc[t] = wmma_bf16(a, b, acc[t]);
            }
        }
#pragma unroll
        for (int t = 0; t < 2; ++t) {
            int col = (w * 2 + t) * 16 + (lane & 15);
#pragma unroll
            for (int r = 0; r < 8; ++r) {
                int m = r + ((lane >> 4) << 3);
                sU[m * 256 + col] = sDelta[m * 256 + col] + acc[t][r] + b2[col];
            }
        }
    }
    __syncthreads();

    // ---- stage 10: layernorm over 256 (half-wave per token) -> sA0 bf16 ----
    {
        int t  = w * 2 + (lane >> 4);
        int li = lane & 15;
        float sum = 0.f, sq = 0.f;
#pragma unroll
        for (int i = 0; i < 16; ++i) {
            float v = sU[t * 256 + li + 16 * i];
            sum += v;
            sq  += v * v;
        }
#pragma unroll
        for (int off = 8; off >= 1; off >>= 1) {
            sum += __shfl_xor(sum, off, 16);
            sq  += __shfl_xor(sq,  off, 16);
        }
        float mean = sum * (1.f / 256.f);
        float var  = sq * (1.f / 256.f) - mean * mean;
        float inv  = rsqrtf(var + 1e-5f);
#pragma unroll
        for (int i = 0; i < 16; ++i) {
            int col = li + 16 * i;
            float v = (sU[t * 256 + col] - mean) * inv * gamma[col] + beta[col];
            sA0[t * 256 + col] = __float2bfloat16(v);
        }
    }
    __syncthreads();

    // ---- stage 11: out = sigmoid(s_norm @ W_bb^T + b_bb) (16x4) ----
    if (w == 0) {
        v8f acc = {};
#pragma unroll
        for (int s = 0; s < 8; ++s) {
            v16bf a = load_a_frag(sA0d, 128, s * 32, lane);
            v16bf b = load_b_frag(pWbb, s, lane);
            acc = wmma_bf16(a, b, acc);
        }
        int n = lane & 15;
        if (n < 4) {
#pragma unroll
            for (int r = 0; r < 8; ++r) {
                int m = r + ((lane >> 4) << 3);
                out[(size_t)(tok0 + m) * 4 + n] = sigmoidf(acc[r] + b_bb[n]);
            }
        }
    }
}

extern "C" void kernel_launch(void* const* d_in, const int* in_sizes, int n_in,
                              void* d_out, int out_size, void* d_ws, size_t ws_size,
                              hipStream_t stream) {
    const float* x0    = (const float*)d_in[0];
    const float* ff    = (const float*)d_in[1];
    const float* h     = (const float*)d_in[2];
    const float* Win   = (const float*)d_in[3];
    const float* Wx    = (const float*)d_in[4];
    const float* Wdt   = (const float*)d_in[5];
    const float* bdt   = (const float*)d_in[6];
    const float* Alog  = (const float*)d_in[7];
    const float* Dvv   = (const float*)d_in[8];
    const float* W1    = (const float*)d_in[9];
    const float* b1    = (const float*)d_in[10];
    const float* W2    = (const float*)d_in[11];
    const float* b2    = (const float*)d_in[12];
    const float* gam   = (const float*)d_in[13];
    const float* bet   = (const float*)d_in[14];
    const float* Wbb   = (const float*)d_in[15];
    const float* bbb   = (const float*)d_in[16];

    unsigned int* ws = (unsigned int*)d_ws;
    size_t o = 0;
    unsigned int* pWin = ws + o; o += (size_t)32 * 8  * 32 * 8;
    unsigned int* pWx  = ws + o; o += (size_t)3  * 8  * 32 * 8;
    unsigned int* pWdt = ws + o; o += (size_t)16 * 1  * 32 * 8;
    unsigned int* pW1  = ws + o; o += (size_t)32 * 8  * 32 * 8;
    unsigned int* pW2  = ws + o; o += (size_t)16 * 16 * 32 * 8;
    unsigned int* pWbb = ws + o; o += (size_t)1  * 8  * 32 * 8;

    {   // pack weights to WMMA-friendly bf16 fragments (L2-resident)
        int t;
        t = 32 * 8 * 32;  pack_wb<<<(t + 255) / 256, 256, 0, stream>>>(Win, 512, 256, 8,  32, pWin);
        t = 3  * 8 * 32;  pack_wb<<<(t + 255) / 256, 256, 0, stream>>>(Wx,  48,  256, 8,  3,  pWx);
        t = 16 * 1 * 32;  pack_wb<<<(t + 255) / 256, 256, 0, stream>>>(Wdt, 256, 16,  1,  16, pWdt);
        t = 32 * 8 * 32;  pack_wb<<<(t + 255) / 256, 256, 0, stream>>>(W1,  512, 256, 8,  32, pW1);
        t = 16 * 16 * 32; pack_wb<<<(t + 255) / 256, 256, 0, stream>>>(W2,  256, 512, 16, 16, pW2);
        t = 1  * 8 * 32;  pack_wb<<<(t + 255) / 256, 256, 0, stream>>>(Wbb, 4,   256, 8,  1,  pWbb);
    }

    float* outp = (float*)d_out;
    float* hnew = outp + (size_t)BTOK * 4;
    mamba_main<<<BTOK / TOK, 256, 0, stream>>>(
        x0, ff, h, bdt, Alog, Dvv, b1, b2, gam, bet, bbb,
        pWin, pWx, pWdt, pW1, pW2, pWbb, outp, hnew);
}
